// TopDownAttentionCompressor_85641647882631
// MI455X (gfx1250) — matrix operand order repaired and checked by reading, not compiled, not run on hardware
//
#include <hip/hip_runtime.h>
#include <hip/hip_bf16.h>
#include <cstdint>

typedef __attribute__((ext_vector_type(16))) __bf16 v16bf;
typedef __attribute__((ext_vector_type(8)))  float  v8f;
typedef union { v16bf v; unsigned u[8]; } v16bf_u;

#define BATCH  16
#define SEQT   4096
#define DMODEL 2048
#define DCOMP  256
#define NHEAD  4
#define NLAT   4
#define DHEAD  64
#define BTROWS (BATCH*SEQT)
#define ASTRIDE 68   /* padded f32 row stride for A tile: 272B -> bank-conflict free */

// ---- f32 -> bf16 round-to-nearest-even (used once, in weight prep) ----
__device__ __forceinline__ unsigned short f2bf(float f) {
  unsigned u = __builtin_bit_cast(unsigned, f);
  unsigned r = (u + 0x7FFFu + ((u >> 16) & 1u)) >> 16;
  return (unsigned short)r;
}

// ---- pack two f32 -> packed bf16x2 (truncation) with one v_perm_b32 ----
__device__ __forceinline__ unsigned pkbf(float a, float b) {
  return __builtin_amdgcn_perm(__builtin_bit_cast(unsigned, b),
                               __builtin_bit_cast(unsigned, a), 0x07060302u);
}

// LDS byte offset of a generic pointer to __shared__ (low 32 bits of aperture addr)
#define LDSOFF(p) ((unsigned)(uintptr_t)(p))

// =====================================================================
// Kernel 0: Q = latent_queries @ Wq + bq   (tiny: [4,256]x[256,256])
// =====================================================================
__global__ __launch_bounds__(256) void q_proj_kernel(
    const float* __restrict__ latq, const float* __restrict__ Wq,
    const float* __restrict__ bq, float* __restrict__ wsQ) {
  int c = threadIdx.x;
  for (int l = 0; l < NLAT; ++l) {
    float acc = bq[c];
    for (int m = 0; m < DCOMP; ++m)
      acc += latq[l * DCOMP + m] * Wq[m * DCOMP + c];
    wsQ[l * DCOMP + c] = acc;
  }
}

// =====================================================================
// Kernel W: one-time convert [Wk|Wv] f32 -> bf16, pre-swizzled into the
// exact WMMA B-fragment layout. Chunk (tileN, ktile) = contiguous 16KB.
//   wsWB[tileN][ktile][ (kfrag*8+nfrag)*512 + lane*16 + e ]
// =====================================================================
__global__ __launch_bounds__(256) void wprep_kernel(
    const float* __restrict__ Wk, const float* __restrict__ Wv,
    unsigned short* __restrict__ wsWB) {
  int gid = blockIdx.x * 256 + threadIdx.x;   // 4 * 2048 * 128 = 1,048,576
  int tileN = gid >> 18;
  int rem   = gid & ((1 << 18) - 1);
  int k = rem >> 7;          // 0..2047
  int n = rem & 127;         // 0..127
  const float* W = (tileN < 2) ? Wk : Wv;
  float f = W[(size_t)k * DCOMP + (tileN & 1) * 128 + n];
  int kin   = k & 31;
  int kfrag = (k >> 5) & 1;
  int nfrag = n >> 4, nin = n & 15;
  int lanei = nin + 16 * ((kin >> 3) & 1);
  int e     = (kin & 7) + 8 * (kin >> 4);
  size_t dst = ((size_t)tileN * 32 + (k >> 6)) * 8192 +
               (kfrag * 8 + nfrag) * 512 + (size_t)lanei * 16 + e;
  wsWB[dst] = f2bf(f);
}

// =====================================================================
// Kernel 1: fused K/V projection GEMM (bf16 WMMA, f32 accum)
//   Double-buffered async global->LDS pipeline (ASYNCcnt):
//     A tile: f32 row-major (padded), bf16-packed in-register per fragment
//     B tile: pre-swizzled bf16, contiguous 16KB copy
//   blockIdx.y: 0,1 -> K (fused Q.K^T logits); 2,3 -> V to workspace
// =====================================================================
__global__ __launch_bounds__(256) void kv_gemm_kernel(
    const float* __restrict__ HS,
    const float* __restrict__ bk, const float* __restrict__ bv,
    const unsigned short* __restrict__ wsWB, const float* __restrict__ wsQ,
    float* __restrict__ wsLogits, float* __restrict__ wsV) {

  // A double buffer (2 x 128 x 68 f32 = 69632B); overlaid by C tile after loop
  __shared__ __align__(32) unsigned char smem[2 * 128 * ASTRIDE * 4];
  __shared__ __align__(32) unsigned short lBs[2][8192];   // B double buffer, 32KB
  __shared__ float lQ[NLAT * DCOMP];
  __shared__ float lBias[128];

  float* lAf = (float*)smem;
  float* lC  = (float*)smem;                 // 128 x 132 after the loop

  const int tid  = threadIdx.x;
  const int lane = tid & 31;
  const int wv   = tid >> 5;
  const int wm   = wv & 3;                   // 4 waves along M (32 rows)
  const int wn   = wv >> 2;                  // 2 waves along N (64 cols)

  const int rowBase  = blockIdx.x * 128;
  const int tileN    = blockIdx.y;
  const bool isK     = (tileN < 2);
  const float* bsel  = isK ? bk : bv;
  const int ncolBase = (tileN & 1) * 128;

  for (int i = tid; i < NLAT * DCOMP; i += 256) lQ[i] = wsQ[i];
  if (tid < 128) lBias[tid] = bsel[ncolBase + tid];

  // issue one k-chunk's async copies (8 A + 4 B b128 per thread = 12)
  auto issue_chunk = [&](int kt, int buf) {
    float* Ab = lAf + buf * (128 * ASTRIDE);
    for (int i = 0; i < 8; ++i) {
      int idx = tid * 8 + i;                 // 2048 b128 for A
      int r = idx >> 4, c4 = idx & 15;
      unsigned l = LDSOFF(Ab + r * ASTRIDE + c4 * 4);
      unsigned long long g =
          (unsigned long long)(HS + (size_t)(rowBase + r) * DMODEL + kt + c4 * 4);
      asm volatile("global_load_async_to_lds_b128 %0, %1, off"
                   :: "v"(l), "v"(g) : "memory");
    }
    const unsigned short* src = wsWB + ((size_t)tileN * 32 + (kt >> 6)) * 8192;
    unsigned lb = LDSOFF(&lBs[buf][0]) + tid * 64;
    unsigned long long gb = (unsigned long long)((const unsigned char*)src + tid * 64);
    for (int j = 0; j < 4; ++j) {
      asm volatile("global_load_async_to_lds_b128 %0, %1, off"
                   :: "v"(lb + (unsigned)(j * 16)), "v"(gb + (unsigned long long)(j * 16))
                   : "memory");
    }
  };

  v8f acc[2][4] = {};
  issue_chunk(0, 0);
  int pb = 0;
  for (int kt = 0; kt < DMODEL; kt += 64) {
    if (kt + 64 < DMODEL) {
      issue_chunk(kt + 64, pb ^ 1);                       // prefetch next chunk
      asm volatile("s_wait_asynccnt 0xc" ::: "memory");   // current chunk landed
    } else {
      asm volatile("s_wait_asynccnt 0x0" ::: "memory");
    }
    __syncthreads();

    const float* Ab = lAf + pb * (128 * ASTRIDE);
    const v16bf* pB = (const v16bf*)&lBs[pb][0];
    for (int kf = 0; kf < 2; ++kf) {
      v16bf_u af[2];
      for (int i = 0; i < 2; ++i) {
        const float* pa = Ab + ((wm * 2 + i) * 16 + (lane & 15)) * ASTRIDE +
                          kf * 32 + 8 * (lane >> 4);
        float4 r0 = *(const float4*)(pa);        // k  0..3  (group 0)
        float4 r1 = *(const float4*)(pa + 4);    // k  4..7
        float4 r2 = *(const float4*)(pa + 16);   // k 16..19 (group 1)
        float4 r3 = *(const float4*)(pa + 20);   // k 20..23
        af[i].u[0] = pkbf(r0.x, r0.y); af[i].u[1] = pkbf(r0.z, r0.w);
        af[i].u[2] = pkbf(r1.x, r1.y); af[i].u[3] = pkbf(r1.z, r1.w);
        af[i].u[4] = pkbf(r2.x, r2.y); af[i].u[5] = pkbf(r2.z, r2.w);
        af[i].u[6] = pkbf(r3.x, r3.y); af[i].u[7] = pkbf(r3.z, r3.w);
      }
      v16bf bfv[4];
      for (int j = 0; j < 4; ++j)
        bfv[j] = pB[(kf * 8 + wn * 4 + j) * 32 + lane];
      for (int i = 0; i < 2; ++i)
        for (int j = 0; j < 4; ++j)
          acc[i][j] = __builtin_amdgcn_wmma_f32_16x16x32_bf16(
              false, af[i].v, false, bfv[j], (short)0, acc[i][j], false, false);
    }
    __syncthreads();   // all reads of buffer pb done before it is refilled
    pb ^= 1;
  }

  // ---- stage C tile to LDS (ISA C/D lane->element mapping) ----
  for (int i = 0; i < 2; ++i)
    for (int j = 0; j < 4; ++j) {
      int mBase = wm * 32 + i * 16;
      int nBase = wn * 64 + j * 16;
      for (int r = 0; r < 8; ++r) {
        int m = mBase + r + 8 * (lane >> 4);
        int n = nBase + (lane & 15);
        lC[m * 132 + n] = acc[i][j][r];
      }
    }
  __syncthreads();

  const int b   = rowBase >> 12;
  const int tg0 = rowBase & (SEQT - 1);

  if (isK) {
    // fused attention logits: Q . K^T for 2 heads, 4 latents, 128 t's
    const int h0 = tileN * 2;
    for (int s = 0; s < 4; ++s) {
      int jid = tid + 256 * s;
      int t  = jid >> 3;
      int hl = (jid >> 2) & 1;
      int l  = jid & 3;
      int h  = h0 + hl;
      float dot = 0.f;
      for (int d = 0; d < DHEAD; ++d)
        dot += lQ[l * DCOMP + h * DHEAD + d] *
               (lC[t * 132 + hl * DHEAD + d] + lBias[hl * DHEAD + d]);
      wsLogits[(((size_t)b * NHEAD + h) * NLAT + l) * SEQT + tg0 + t] = dot * 0.125f;
    }
  } else {
    // V + bias -> workspace, coalesced float4 stores
    for (int i = 0; i < 16; ++i) {
      int idx = i * 256 + tid;
      int r = idx >> 5, c4 = idx & 31;
      int n = c4 * 4;
      float4 o;
      o.x = lC[r * 132 + n + 0] + lBias[n + 0];
      o.y = lC[r * 132 + n + 1] + lBias[n + 1];
      o.z = lC[r * 132 + n + 2] + lBias[n + 2];
      o.w = lC[r * 132 + n + 3] + lBias[n + 3];
      *(float4*)(wsV + (size_t)(rowBase + r) * DCOMP + ncolBase + n) = o;
    }
  }
}

// =====================================================================
// Kernel 2: masked softmax + attn@V + Wo proj + residual + LayerNorm
// =====================================================================
__global__ __launch_bounds__(256) void attn_out_kernel(
    const float* __restrict__ wsLogits, const float* __restrict__ wsV,
    const int* __restrict__ mask, const float* __restrict__ latq,
    const float* __restrict__ Wo, const float* __restrict__ bo,
    const float* __restrict__ gamma, const float* __restrict__ beta,
    float* __restrict__ out) {

  __shared__ float prob[NLAT][SEQT];       // 64 KB
  __shared__ float red[256];
  __shared__ float mrow[NLAT], srow[NLAT];
  __shared__ float outSh[NLAT * DCOMP];
  __shared__ float part[4][NLAT][DHEAD];

  const int b = blockIdx.x;
  const int tid = threadIdx.x;
  const int* mb = mask + (size_t)b * SEQT;

  for (int i = tid; i < NLAT * DCOMP; i += 256) outSh[i] = 0.f;

  for (int h = 0; h < NHEAD; ++h) {
    const float* lg = wsLogits + ((size_t)(b * NHEAD + h) * NLAT) * SEQT;
    for (int l = 0; l < NLAT; ++l) {
      float mx = -1e30f;
      for (int t = tid; t < SEQT; t += 256) {
        float x = (mb[t] == 0) ? -1e30f : lg[l * SEQT + t];
        mx = fmaxf(mx, x);
      }
      red[tid] = mx; __syncthreads();
      for (int s = 128; s > 0; s >>= 1) { if (tid < s) red[tid] = fmaxf(red[tid], red[tid + s]); __syncthreads(); }
      float M = red[0]; __syncthreads();
      float sm = 0.f;
      for (int t = tid; t < SEQT; t += 256)
        if (mb[t] != 0) sm += __expf(lg[l * SEQT + t] - M);
      red[tid] = sm; __syncthreads();
      for (int s = 128; s > 0; s >>= 1) { if (tid < s) red[tid] += red[tid + s]; __syncthreads(); }
      if (tid == 0) { mrow[l] = M; srow[l] = fmaxf(red[0], 1e-30f); }
      __syncthreads();
    }
    for (int l = 0; l < NLAT; ++l) {
      float M = mrow[l], inv = 1.f / srow[l];
      for (int t = tid; t < SEQT; t += 256)
        prob[l][t] = (mb[t] == 0) ? 0.f : __expf(lg[l * SEQT + t] - M) * inv;
    }
    __syncthreads();
    int d = tid & 63, g = tid >> 6;
    float a0 = 0.f, a1 = 0.f, a2 = 0.f, a3 = 0.f;
    for (int t = g * 1024; t < (g + 1) * 1024; ++t) {
      float v = wsV[((size_t)b * SEQT + t) * DCOMP + h * DHEAD + d];
      a0 += prob[0][t] * v; a1 += prob[1][t] * v;
      a2 += prob[2][t] * v; a3 += prob[3][t] * v;
    }
    part[g][0][d] = a0; part[g][1][d] = a1; part[g][2][d] = a2; part[g][3][d] = a3;
    __syncthreads();
    if (g == 0)
      for (int l = 0; l < NLAT; ++l)
        outSh[l * DCOMP + h * DHEAD + d] =
            part[0][l][d] + part[1][l][d] + part[2][l][d] + part[3][l][d];
    __syncthreads();
  }

  int c = tid;
  float y[NLAT];
  for (int l = 0; l < NLAT; ++l) {
    float a = bo[c];
    for (int m = 0; m < DCOMP; ++m) a += outSh[l * DCOMP + m] * Wo[m * DCOMP + c];
    y[l] = a + latq[l * DCOMP + c];
  }
  for (int l = 0; l < NLAT; ++l) {
    red[tid] = y[l]; __syncthreads();
    for (int s = 128; s > 0; s >>= 1) { if (tid < s) red[tid] += red[tid + s]; __syncthreads(); }
    float mu = red[0] * (1.f / DCOMP); __syncthreads();
    float dv = y[l] - mu;
    red[tid] = dv * dv; __syncthreads();
    for (int s = 128; s > 0; s >>= 1) { if (tid < s) red[tid] += red[tid + s]; __syncthreads(); }
    float var = red[0] * (1.f / DCOMP); __syncthreads();
    out[(size_t)b * (NLAT * DCOMP) + l * DCOMP + c] =
        dv * rsqrtf(var + 1e-5f) * gamma[c] + beta[c];
  }
}

// =====================================================================
extern "C" void kernel_launch(void* const* d_in, const int* in_sizes, int n_in,
                              void* d_out, int out_size, void* d_ws, size_t ws_size,
                              hipStream_t stream) {
  (void)in_sizes; (void)n_in; (void)out_size; (void)ws_size;
  const float* HS   = (const float*)d_in[0];
  const int*   mask = (const int*)d_in[1];
  const float* latq = (const float*)d_in[2];
  const float* Wq   = (const float*)d_in[3];
  const float* bq   = (const float*)d_in[4];
  const float* Wk   = (const float*)d_in[5];
  const float* bk   = (const float*)d_in[6];
  const float* Wv   = (const float*)d_in[7];
  const float* bv   = (const float*)d_in[8];
  const float* Wo   = (const float*)d_in[9];
  const float* bo   = (const float*)d_in[10];
  const float* gamma= (const float*)d_in[11];
  const float* beta = (const float*)d_in[12];
  float* out = (float*)d_out;

  float* ws       = (float*)d_ws;
  float* wsQ      = ws;                                   // 1,024 f32
  float* wsLogits = ws + 1024;                            // 1,048,576 f32
  float* wsV      = ws + 1024 + 1048576;                  // 16,777,216 f32
  unsigned short* wsWB =
      (unsigned short*)(ws + 1024 + 1048576 + 16777216);  // 1,048,576 bf16

  q_proj_kernel<<<1, 256, 0, stream>>>(latq, Wq, bq, wsQ);
  wprep_kernel<<<4096, 256, 0, stream>>>(Wk, Wv, wsWB);
  kv_gemm_kernel<<<dim3(BTROWS / 128, 4), 256, 0, stream>>>(
      HS, bk, bv, wsWB, wsQ, wsLogits, wsV);
  attn_out_kernel<<<BATCH, 256, 0, stream>>>(
      wsLogits, wsV, mask, latq, Wo, bo, gamma, beta, out);
}
